// ASTEncoder_20864951124525
// MI455X (gfx1250) — matrix operand order repaired and checked by reading, not compiled, not run on hardware
//
#include <hip/hip_runtime.h>
#include <hip/hip_bf16.h>
#include <stdint.h>

// ---------------------------------------------------------------------------
// AST tree-GRU encoder for MI455X (gfx1250).
// Dense math on v_wmma_f32_16x16x32_bf16; tile staging uses the CDNA5 async
// global->LDS path (global_load_async_to_lds_b128, ASYNCcnt) with double
// buffering so WMMA overlaps the next tile's fill. wave32, 8-wave blocks.
// ---------------------------------------------------------------------------

#define NNODES 32768
#define HDIM   256
#define NROUNDS 6

typedef __bf16 bf16_t;
typedef __attribute__((ext_vector_type(16))) __bf16 v16bf;
typedef __attribute__((ext_vector_type(8)))  float  v8f;

struct bf16x16 { uint4 lo, hi; };

__device__ __forceinline__ bf16_t f2b(float x) {
  unsigned u = __float_as_uint(x);
  unsigned r = (u + 0x7FFFu + ((u >> 16) & 1u)) >> 16;   // round-nearest-even
  unsigned short hs = (unsigned short)r;
  return __builtin_bit_cast(bf16_t, hs);
}

// Generic pointer to LDS byte offset (LDS aperture: low 32 bits give the LDS offset)
__device__ __forceinline__ unsigned lds_off(const void* p) {
  return (unsigned)(uintptr_t)p;
}
// Async 16B copy global -> LDS (ASYNCcnt-tracked, no VGPR round trip)
__device__ __forceinline__ void async_cp16(unsigned ldsDst, const void* gsrc) {
  asm volatile("global_load_async_to_lds_b128 %0, %1, off"
               :: "v"(ldsDst), "v"(gsrc) : "memory");
}
__device__ __forceinline__ void wait_async0() {
  asm volatile("s_wait_asynccnt 0x0" ::: "memory");
}

// A fragment (16x32 bf16): lanes 0-15 hold row=lane, K {0..7,16..23};
// lanes 16-31 hold row=lane-16, K {8..15,24..31}. rowPtr -> 32 contiguous K.
__device__ __forceinline__ v16bf load_frag_a(const bf16_t* rowPtr, int half) {
  bf16x16 t;
  t.lo = *(const uint4*)(rowPtr + half * 8);
  t.hi = *(const uint4*)(rowPtr + half * 8 + 16);
  return __builtin_bit_cast(v16bf, t);
}
// B fragment (32x16 bf16, K-contiguous column storage Wt[col][k]):
// lanes 0-15 hold K {0..7,16..23} of column=lane, lanes 16-31 hold K {8..15,24..31}.
__device__ __forceinline__ v16bf load_frag_b(const bf16_t* colPtr, int half) {
  bf16x16 t;
  t.lo = *(const uint4*)(colPtr + half * 8);
  t.hi = *(const uint4*)(colPtr + half * 8 + 16);
  return __builtin_bit_cast(v16bf, t);
}

__device__ __forceinline__ v8f wmma_bf16(v16bf a, v16bf b, v8f c) {
  return __builtin_amdgcn_wmma_f32_16x16x32_bf16(false, a, false, b,
                                                 (short)0, c, false, false);
}

// ---------------------------------------------------------------------------
// GEMM: C[M,Ncols] = A[M,K](bf16) * Wt[Ncols,K](bf16)^T + bias.
// Block tile 256x64, 8 waves; each wave: 32 rows x 64 cols = 8 WMMA/k-step
// (2 A frags x 4 B frags -> 1.5 ds_load_b128 per WMMA). Double-buffered
// async global->LDS staging.
// ---------------------------------------------------------------------------
__global__ __launch_bounds__(256) void gemm_bf16(
    const bf16_t* __restrict__ A, const bf16_t* __restrict__ Wt,
    const float* __restrict__ bias, void* __restrict__ C, int outF32,
    int K, int Ncols)
{
  __shared__ __align__(16) bf16_t lA[2][256][32];   // 2 x 16KB
  __shared__ __align__(16) bf16_t lB[2][64][32];    // 2 x 4KB
  const int tid = threadIdx.x, wave = tid >> 5, lane = tid & 31;
  const int half = lane >> 4, r16 = lane & 15;
  const int m0 = blockIdx.x * 256, n0 = blockIdx.y * 64;

  auto load_tiles = [&](int k0, int buf) {
    // A tile: 256 rows x 64B = 1024 16B chunks; 4 per thread.
    #pragma unroll
    for (int c = 0; c < 4; ++c) {
      int ci = tid + c * 256;
      int row = ci >> 2, off = ci & 3;
      async_cp16(lds_off(&lA[buf][row][0]) + off * 16,
                 A + (size_t)(m0 + row) * K + k0 + off * 8);
    }
    // B tile: 64 rows x 64B = 256 chunks; 1 per thread.
    { int row = tid >> 2, off = tid & 3;
      async_cp16(lds_off(&lB[buf][row][0]) + off * 16,
                 Wt + (size_t)(n0 + row) * K + k0 + off * 8); }
  };

  v8f acc[2][4] = {};
  const int nk = K / 32;
  load_tiles(0, 0);
  wait_async0();
  __syncthreads();
  for (int ki = 0; ki < nk; ++ki) {
    const int buf = ki & 1;
    if (ki + 1 < nk) load_tiles((ki + 1) * 32, buf ^ 1);   // overlap fill
    v16bf a0 = load_frag_a(&lA[buf][wave * 32 + r16][0], half);
    v16bf a1 = load_frag_a(&lA[buf][wave * 32 + 16 + r16][0], half);
    #pragma unroll
    for (int nt = 0; nt < 4; ++nt) {
      v16bf bfr = load_frag_b(&lB[buf][nt * 16 + r16][0], half);
      acc[0][nt] = wmma_bf16(a0, bfr, acc[0][nt]);
      acc[1][nt] = wmma_bf16(a1, bfr, acc[1][nt]);
    }
    wait_async0();          // own async writes to buf^1 done
    __syncthreads();        // everyone's writes visible / reads of buf done
  }
  // C/D layout: VGPR r -> row r (lanes 0-15) / r+8 (lanes 16-31), col=lane&15
  const int colBase = n0 + r16;
  #pragma unroll
  for (int rg = 0; rg < 2; ++rg) {
    const int rowBase = m0 + wave * 32 + rg * 16 + 8 * half;
    #pragma unroll
    for (int nt = 0; nt < 4; ++nt) {
      int col = colBase + nt * 16;
      float bv = bias ? bias[col] : 0.0f;
      #pragma unroll
      for (int r = 0; r < 8; ++r) {
        float v = acc[rg][nt][r] + bv;
        size_t o = (size_t)(rowBase + r) * Ncols + col;
        if (outF32) ((float*)C)[o] = v; else ((bf16_t*)C)[o] = f2b(v);
      }
    }
  }
}

// ---------------------------------------------------------------------------
// Fused GRU for stack s (blockIdx.z): six gate accumulators per 16-row group
// held in registers (12 WMMA per wave per k-step, B frags reused across the
// two row groups), gate nonlinearity in-register, masked store by category.
// grid = (N/32, 2, 3), block = 256 (8 waves x 16 cols each).
// ---------------------------------------------------------------------------
__global__ __launch_bounds__(256) void gru_fused(
    const bf16_t* __restrict__ X, const bf16_t* __restrict__ Hb,
    const float* __restrict__ h,
    const bf16_t* __restrict__ WtIh, const bf16_t* __restrict__ WtHh,  // [3*3*256][256]
    const float* __restrict__ bih, const float* __restrict__ bhh,      // [3][768]
    const unsigned char* __restrict__ catG, float* __restrict__ out)
{
  __shared__ __align__(16) bf16_t lX[32][32];   // 2KB
  __shared__ __align__(16) bf16_t lH[32][32];   // 2KB
  const int tid = threadIdx.x, wave = tid >> 5, lane = tid & 31;
  const int half = lane >> 4, r16 = lane & 15;
  const int m0 = blockIdx.x * 32;
  const int s  = blockIdx.z;
  const int jb = blockIdx.y * 128 + wave * 16;   // column tile base in [0,256)
  v8f g[2][6] = {};                               // [rowgroup][ir,iz,ig,hr,hz,hg]
  for (int k0 = 0; k0 < HDIM; k0 += 32) {
    // 32 rows x 64B = 128 chunks per tile; threads 0..127 -> X, 128..255 -> H
    { int t = tid & 127, row = t >> 2, off = t & 3;
      if (tid < 128)
        async_cp16(lds_off(&lX[row][0]) + off * 16,
                   X + (size_t)(m0 + row) * HDIM + k0 + off * 8);
      else
        async_cp16(lds_off(&lH[row][0]) + off * 16,
                   Hb + (size_t)(m0 + row) * HDIM + k0 + off * 8); }
    wait_async0();
    __syncthreads();
    v16bf ax0 = load_frag_a(&lX[r16][0], half);
    v16bf ax1 = load_frag_a(&lX[16 + r16][0], half);
    v16bf ah0 = load_frag_a(&lH[r16][0], half);
    v16bf ah1 = load_frag_a(&lH[16 + r16][0], half);
    #pragma unroll
    for (int gg = 0; gg < 3; ++gg) {
      const bf16_t* pi = WtIh + ((size_t)((s * 3 + gg) * HDIM + jb + r16)) * HDIM + k0;
      const bf16_t* ph = WtHh + ((size_t)((s * 3 + gg) * HDIM + jb + r16)) * HDIM + k0;
      v16bf bi = load_frag_b(pi, half);   // weights stay hot in L2
      v16bf bh = load_frag_b(ph, half);
      g[0][gg]     = wmma_bf16(ax0, bi, g[0][gg]);
      g[1][gg]     = wmma_bf16(ax1, bi, g[1][gg]);
      g[0][gg + 3] = wmma_bf16(ah0, bh, g[0][gg + 3]);
      g[1][gg + 3] = wmma_bf16(ah1, bh, g[1][gg + 3]);
    }
    __syncthreads();
  }
  const int col = jb + r16;
  const float bir = bih[s * 768 + col],       bhr = bhh[s * 768 + col];
  const float biz = bih[s * 768 + 256 + col], bhz = bhh[s * 768 + 256 + col];
  const float big = bih[s * 768 + 512 + col], bhg = bhh[s * 768 + 512 + col];
  #pragma unroll
  for (int rg = 0; rg < 2; ++rg) {
    const int rowBase = m0 + rg * 16 + 8 * half;
    #pragma unroll
    for (int r = 0; r < 8; ++r) {
      int row = rowBase + r;
      if ((int)catG[row] != s) continue;           // this stack owns this row?
      float hv = h[(size_t)row * HDIM + col];
      float rr = 1.0f / (1.0f + __expf(-(g[rg][0][r] + bir + g[rg][3][r] + bhr)));
      float zz = 1.0f / (1.0f + __expf(-(g[rg][1][r] + biz + g[rg][4][r] + bhz)));
      float nn = tanhf(g[rg][2][r] + big + rr * (g[rg][5][r] + bhg));
      out[(size_t)row * HDIM + col] = (1.0f - zz) * nn + zz * hv;
    }
  }
}

// Residual + LayerNorm, in-place into h. One wave per row (8 rows/block).
__global__ __launch_bounds__(256) void ln_resid(
    const float* __restrict__ gout, float* __restrict__ h,
    const float* __restrict__ lng, const float* __restrict__ lnb)
{
  const int wave = threadIdx.x >> 5, lane = threadIdx.x & 31;
  const int row = blockIdx.x * 8 + wave;
  const size_t base = (size_t)row * HDIM + lane * 8;
  float v[8]; float s = 0.0f;
  #pragma unroll
  for (int c = 0; c < 8; ++c) { v[c] = gout[base + c] + h[base + c]; s += v[c]; }
  #pragma unroll
  for (int o = 16; o > 0; o >>= 1) s += __shfl_xor(s, o, 32);
  const float mu = s * (1.0f / HDIM);
  float q = 0.0f;
  #pragma unroll
  for (int c = 0; c < 8; ++c) { float d = v[c] - mu; q += d * d; }
  #pragma unroll
  for (int o = 16; o > 0; o >>= 1) q += __shfl_xor(q, o, 32);
  const float rs = rsqrtf(q * (1.0f / HDIM) + 1e-5f);
  #pragma unroll
  for (int c = 0; c < 8; ++c)
    h[base + c] = (v[c] - mu) * rs * lng[lane * 8 + c] + lnb[lane * 8 + c];
}

// Child aggregation (complete binary tree: no segment_sum needed) + bf16 casts.
__global__ void agg_build(const float* __restrict__ h,
                          const int* __restrict__ child0, const int* __restrict__ child1,
                          const int* __restrict__ nch,
                          bf16_t* __restrict__ Xcomm, bf16_t* __restrict__ Xbin,
                          bf16_t* __restrict__ Hb)
{
  size_t idx = (size_t)blockIdx.x * 256 + threadIdx.x;  // over N*H
  int i = (int)(idx >> 8), c = (int)(idx & 255);
  int nc = nch[i];
  float h0 = h[(size_t)child0[i] * HDIM + c];
  float h1 = h[(size_t)child1[i] * HDIM + c];
  float s0 = (nc >= 1) ? h0 : 0.0f;
  float s1 = (nc == 2) ? h1 : 0.0f;
  float mean = (s0 + s1) / fmaxf((float)nc, 1.0f);
  float smax = (nc == 0) ? 0.0f : ((nc == 2) ? fmaxf(h0, h1) : h0);
  Xcomm[(size_t)i * 512 + c]       = f2b(mean);
  Xcomm[(size_t)i * 512 + 256 + c] = f2b(smax);
  Xbin [(size_t)i * 512 + c]       = f2b(h0);
  Xbin [(size_t)i * 512 + 256 + c] = f2b(h1);
  Hb[idx] = f2b(h[idx]);
}

// Initial embedding concat -> bf16 [N,512]
__global__ void embed_build(const int* __restrict__ nt, const int* __restrict__ ci,
                            const int* __restrict__ vi,
                            const float* __restrict__ te, const float* __restrict__ ce,
                            const float* __restrict__ ve, bf16_t* __restrict__ X0)
{
  size_t idx = (size_t)blockIdx.x * 256 + threadIdx.x;  // over N*H
  int i = (int)(idx >> 8), c = (int)(idx & 255);
  int t = nt[i];
  float tv = te[t * HDIM + c];
  float vv = (t == 3) ? ce[ci[i] * HDIM + c] : ((t == 4) ? ve[vi[i] * HDIM + c] : 0.0f);
  X0[(size_t)i * 512 + c]       = f2b(tv);
  X0[(size_t)i * 512 + 256 + c] = f2b(vv);
}

// Static per-node categories: GRU stack (0 leaf /1 comm /2 bin) and agg source.
__global__ void cat_build(const int* __restrict__ nt, const int* __restrict__ nch,
                          unsigned char* __restrict__ catG, unsigned char* __restrict__ catA)
{
  int i = blockIdx.x * 256 + threadIdx.x;
  if (i >= NNODES) return;
  int t = nt[i], nc = nch[i];
  catG[i] = (nc == 0) ? 0 : ((t == 2 && nc == 2) ? 2 : 1);
  unsigned char ca;
  if (nc == 0) ca = 0;
  else if (t == 2 && nc == 2) ca = 2;
  else if ((t == 0 || t == 1) && nc > 0) ca = 1;
  else ca = 3;                                     // mean fallthrough
  catA[i] = ca;
}

// agg select -> X bf16 [N,H]
__global__ void agg_select(const bf16_t* __restrict__ aggc, const bf16_t* __restrict__ aggb,
                           const bf16_t* __restrict__ Xcomm,
                           const unsigned char* __restrict__ catA, bf16_t* __restrict__ X)
{
  size_t idx = (size_t)blockIdx.x * 256 + threadIdx.x;  // over N*H
  int i = (int)(idx >> 8), c = (int)(idx & 255);
  unsigned char ca = catA[i];
  bf16_t v;
  if (ca == 0)      v = f2b(0.0f);
  else if (ca == 1) v = aggc[idx];
  else if (ca == 2) v = aggb[idx];
  else              v = Xcomm[(size_t)i * 512 + c];     // mean (bf16)
  X[idx] = v;
}

// Weight prep: fp32 W[K,Ncols] -> bf16 Wt[Ncols,K] (K-contiguous for B frags)
__global__ void wt_build(const float* __restrict__ W, bf16_t* __restrict__ Wt,
                         int K, int Ncols)
{
  size_t idx = (size_t)blockIdx.x * 256 + threadIdx.x;
  if (idx >= (size_t)K * Ncols) return;
  int j = (int)(idx % Ncols), k = (int)(idx / Ncols);
  Wt[(size_t)j * K + k] = f2b(W[(size_t)k * Ncols + j]);
}

// global_ctx = h[0] @ W_glob + b_glob (tiny, fp32 VALU)
__global__ void glob_kernel(const float* __restrict__ h, const float* __restrict__ Wg,
                            const float* __restrict__ bg, float* __restrict__ gctx)
{
  int j = threadIdx.x;
  float acc = bg[j];
  for (int k = 0; k < HDIM; ++k) acc += h[k] * Wg[k * HDIM + j];
  gctx[j] = acc;
}

// node_emb = concat(h, broadcast(global_ctx))
__global__ void out_kernel(const float* __restrict__ h, const float* __restrict__ gctx,
                           float* __restrict__ out)
{
  size_t idx = (size_t)blockIdx.x * 256 + threadIdx.x;  // over N*2H
  size_t i = idx >> 9; int c = (int)(idx & 511);
  out[idx] = (c < HDIM) ? h[i * HDIM + c] : gctx[c - HDIM];
}

// ---------------------------------------------------------------------------
extern "C" void kernel_launch(void* const* d_in, const int* in_sizes, int n_in,
                              void* d_out, int out_size, void* d_ws, size_t ws_size,
                              hipStream_t stream)
{
  (void)in_sizes; (void)n_in; (void)out_size; (void)ws_size;
  const int*   node_type = (const int*)d_in[0];
  const int*   coeff_idx = (const int*)d_in[1];
  const int*   var_idx   = (const int*)d_in[2];
  const int*   child0    = (const int*)d_in[5];
  const int*   child1    = (const int*)d_in[6];
  const int*   nch       = (const int*)d_in[7];
  const float* type_emb  = (const float*)d_in[8];
  const float* coeff_emb = (const float*)d_in[9];
  const float* var_emb   = (const float*)d_in[10];
  const float* W_init = (const float*)d_in[11]; const float* b_init = (const float*)d_in[12];
  const float* W_comm = (const float*)d_in[13]; const float* b_comm = (const float*)d_in[14];
  const float* W_bin  = (const float*)d_in[15]; const float* b_bin  = (const float*)d_in[16];
  const float* Wih    = (const float*)d_in[17]; const float* Whh    = (const float*)d_in[18];
  const float* bih    = (const float*)d_in[19]; const float* bhh    = (const float*)d_in[20];
  const float* lng    = (const float*)d_in[21]; const float* lnb    = (const float*)d_in[22];
  const float* Wglob  = (const float*)d_in[23]; const float* bglob  = (const float*)d_in[24];

  char* ws = (char*)d_ws;
  size_t off = 0;
  auto take = [&](size_t bytes) -> char* {
    char* p = ws + off;
    off += (bytes + 255) & ~(size_t)255;
    return p;
  };
  float*  h     = (float*) take((size_t)NNODES * HDIM * 4);   // 32 MB
  bf16_t* Hb    = (bf16_t*)take((size_t)NNODES * HDIM * 2);   // 16 MB
  bf16_t* Xcomm = (bf16_t*)take((size_t)NNODES * 512 * 2);    // 32 MB
  bf16_t* Xbin  = (bf16_t*)take((size_t)NNODES * 512 * 2);    // 32 MB
  bf16_t* AggC  = (bf16_t*)take((size_t)NNODES * HDIM * 2);   // 16 MB
  bf16_t* AggB  = (bf16_t*)take((size_t)NNODES * HDIM * 2);   // 16 MB
  bf16_t* X     = (bf16_t*)take((size_t)NNODES * HDIM * 2);   // 16 MB
  float*  gout  = (float*) take((size_t)NNODES * HDIM * 4);   // 32 MB
  unsigned char* catG = (unsigned char*)take(NNODES);
  unsigned char* catA = (unsigned char*)take(NNODES);
  bf16_t* WtInit = (bf16_t*)take((size_t)512 * 256 * 2);
  bf16_t* WtComm = (bf16_t*)take((size_t)512 * 256 * 2);
  bf16_t* WtBin  = (bf16_t*)take((size_t)512 * 256 * 2);
  bf16_t* WtIh   = (bf16_t*)take((size_t)3 * 768 * 256 * 2);
  bf16_t* WtHh   = (bf16_t*)take((size_t)3 * 768 * 256 * 2);

  const dim3 blk(256);
  const int nhBlocks = NNODES * HDIM / 256;        // 32768

  // -- one-time (per launch) weight prep + categories --
  wt_build<<<(512 * 256 + 255) / 256, blk, 0, stream>>>(W_init, WtInit, 512, 256);
  wt_build<<<(512 * 256 + 255) / 256, blk, 0, stream>>>(W_comm, WtComm, 512, 256);
  wt_build<<<(512 * 256 + 255) / 256, blk, 0, stream>>>(W_bin,  WtBin,  512, 256);
  for (int s = 0; s < 3; ++s) {
    wt_build<<<(256 * 768 + 255) / 256, blk, 0, stream>>>(
        Wih + (size_t)s * 256 * 768, WtIh + (size_t)s * 768 * 256, 256, 768);
    wt_build<<<(256 * 768 + 255) / 256, blk, 0, stream>>>(
        Whh + (size_t)s * 256 * 768, WtHh + (size_t)s * 768 * 256, 256, 768);
  }
  cat_build<<<(NNODES + 255) / 256, blk, 0, stream>>>(node_type, nch, catG, catA);

  // -- initial states: h = concat(te, ve) @ W_init + b_init --
  embed_build<<<nhBlocks, blk, 0, stream>>>(node_type, coeff_idx, var_idx,
                                            type_emb, coeff_emb, var_emb, Xcomm);
  gemm_bf16<<<dim3(NNODES / 256, HDIM / 64), blk, 0, stream>>>(
      Xcomm, WtInit, b_init, h, /*outF32=*/1, /*K=*/512, /*Ncols=*/HDIM);

  // -- message-passing rounds --
  for (int rnd = 0; rnd < NROUNDS; ++rnd) {
    agg_build<<<nhBlocks, blk, 0, stream>>>(h, child0, child1, nch, Xcomm, Xbin, Hb);
    gemm_bf16<<<dim3(NNODES / 256, HDIM / 64), blk, 0, stream>>>(
        Xcomm, WtComm, b_comm, AggC, /*outF32=*/0, 512, HDIM);
    gemm_bf16<<<dim3(NNODES / 256, HDIM / 64), blk, 0, stream>>>(
        Xbin, WtBin, b_bin, AggB, /*outF32=*/0, 512, HDIM);
    agg_select<<<nhBlocks, blk, 0, stream>>>(AggC, AggB, Xcomm, catA, X);
    gru_fused<<<dim3(NNODES / 32, 2, 3), blk, 0, stream>>>(
        X, Hb, h, WtIh, WtHh, bih, bhh, catG, gout);
    ln_resid<<<NNODES / 8, blk, 0, stream>>>(gout, h, lng, lnb);
  }

  // -- outputs --
  float* outp = (float*)d_out;
  float* gctx = outp + (size_t)NNODES * 512;       // global_ctx lives after node_emb
  glob_kernel<<<1, blk, 0, stream>>>(h, Wglob, bglob, gctx);
  out_kernel<<<NNODES * 512 / 256, blk, 0, stream>>>(h, gctx, outp);
}
